// CompositionalGraphReasoning_19937238188709
// MI455X (gfx1250) — matrix operand order repaired and checked by reading, not compiled, not run on hardware
//
#include <hip/hip_runtime.h>

// ---------------------------------------------------------------------------
// CDNA5 / gfx1250 implementation of CompositionalGraphReasoning.
// Heavy GEMMs + attention use v_wmma_f32_16x16x32_bf16 (wave32, 8-wave blocks).
// GEMM is double-buffered with GLOBAL_LOAD_ASYNC_TO_LDS (ASYNCcnt-tracked).
// ---------------------------------------------------------------------------

typedef __bf16 bf16;
typedef bf16  v16bf __attribute__((ext_vector_type(16)));
typedef float v8f   __attribute__((ext_vector_type(8)));
typedef unsigned int u32x4 __attribute__((ext_vector_type(4)));
typedef int v4i_t __attribute__((ext_vector_type(4)));

union Frag16 { v16bf v; u32x4 q[2]; };
union Acc8   { v8f  v; float f[8]; };

#define WMMA_BF16(ACC, A, B)                                                  \
  (ACC) = __builtin_amdgcn_wmma_f32_16x16x32_bf16(false, (A), false, (B),     \
                                                  (short)0, (ACC), false, false)

#if defined(__gfx1250__) &&                                                   \
    __has_builtin(__builtin_amdgcn_global_load_async_to_lds_b128) &&          \
    __has_builtin(__builtin_amdgcn_s_wait_asynccnt)
#define USE_ASYNC_LDS 1
typedef __attribute__((address_space(1))) v4i_t* gas_ptr_t;  // global int4*
typedef __attribute__((address_space(3))) v4i_t* las_ptr_t;  // LDS int4*
#else
#define USE_ASYNC_LDS 0
#endif

// Stage one 16-byte chunk global -> LDS (async DMA when available).
__device__ __forceinline__ void stage_b128(const unsigned short* g,
                                           unsigned short* l) {
#if USE_ASYNC_LDS
  __builtin_amdgcn_global_load_async_to_lds_b128(
      (gas_ptr_t)(unsigned long long)g, (las_ptr_t)(unsigned long long)l, 0, 0);
#else
  *(u32x4*)l = *(const u32x4*)g;
#endif
}
__device__ __forceinline__ void wait_async_all() {
#if USE_ASYNC_LDS
  __builtin_amdgcn_s_wait_asynccnt(0);
#endif
}

__device__ __forceinline__ unsigned short f32_to_bf16(float f) {
  unsigned u = __float_as_uint(f);
  unsigned r = (u + 0x7FFFu + ((u >> 16) & 1u)) >> 16;
  return (unsigned short)r;
}
__device__ __forceinline__ float bf16_to_f32(unsigned short h) {
  return __uint_as_float(((unsigned)h) << 16);
}

__device__ __forceinline__ float block_sum(float v, float* red) {
  int tid = threadIdx.x;
  red[tid] = v;
  __syncthreads();
  for (int s = 128; s > 0; s >>= 1) {
    if (tid < s) red[tid] += red[tid + s];
    __syncthreads();
  }
  float r = red[0];
  __syncthreads();
  return r;
}

// ---------------------------------------------------------------------------
// fp32 -> bf16 conversion
// ---------------------------------------------------------------------------
__global__ __launch_bounds__(256) void cvt_f32_bf16_kernel(
    const float* __restrict__ src, unsigned short* __restrict__ dst, int n) {
  int i = blockIdx.x * blockDim.x + threadIdx.x;
  int stride = gridDim.x * blockDim.x;
  for (; i < n; i += stride) dst[i] = f32_to_bf16(src[i]);
}

// ---------------------------------------------------------------------------
// WMMA bf16 GEMM:  out[M,N] = A[M,K] * W[N,K]^T + bias
// Block tile 128x128, 8 waves; each wave 32x64 (2x4 16x16 accumulators).
// Double-buffered LDS, async global->LDS staging of the next K-tile.
// ---------------------------------------------------------------------------
__global__ __launch_bounds__(256) void gemm_bf16_kernel(
    const unsigned short* __restrict__ A,   // [M,K] bf16 row-major
    const unsigned short* __restrict__ W,   // [N,K] bf16 row-major (i.e. B^T)
    const float* __restrict__ bias,         // [N] or null
    float* __restrict__ outF,               // [M,N] f32 (or null)
    unsigned short* __restrict__ outB,      // [M,N] bf16 (or null)
    int M, int N, int K) {
  __shared__ alignas(16) unsigned short As[2][128][40];  // 32 K + 8 pad
  __shared__ alignas(16) unsigned short Bs[2][128][40];

  int tid = threadIdx.x;
  int w = tid >> 5, lane = tid & 31, hi = lane >> 4, ln = lane & 15;
  int wm = w & 3, wn = w >> 2;           // 4 row-tiles(32) x 2 col-groups(64)
  int m0 = blockIdx.x * 128, n0 = blockIdx.y * 128;

  Acc8 acc[2][4];
#pragma unroll
  for (int at = 0; at < 2; ++at)
#pragma unroll
    for (int bt = 0; bt < 4; ++bt)
#pragma unroll
      for (int j = 0; j < 8; ++j) acc[at][bt].f[j] = 0.f;

  int srow = tid >> 2, sseg = tid & 3;   // 64 rows x 4 b128-segments, x2 halves

#define STAGE_TILE(KT, BUF)                                                    \
  do {                                                                         \
    const unsigned short* ga = A + (size_t)(m0 + srow) * K + (KT) * 32 + sseg * 8; \
    stage_b128(ga, &As[BUF][srow][sseg * 8]);                                  \
    stage_b128(ga + (size_t)64 * K, &As[BUF][srow + 64][sseg * 8]);            \
    const unsigned short* gb = W + (size_t)(n0 + srow) * K + (KT) * 32 + sseg * 8; \
    stage_b128(gb, &Bs[BUF][srow][sseg * 8]);                                  \
    stage_b128(gb + (size_t)64 * K, &Bs[BUF][srow + 64][sseg * 8]);            \
  } while (0)

  int nk = K >> 5;
  STAGE_TILE(0, 0);
  for (int kt = 0; kt < nk; ++kt) {
    int buf = kt & 1;
    wait_async_all();        // tile kt (this wave's share) has landed
    __syncthreads();         // everyone's share landed; prev reads of buf done
    if (kt + 1 < nk) STAGE_TILE(kt + 1, buf ^ 1);  // overlap with compute

    Frag16 af[2];
#pragma unroll
    for (int at = 0; at < 2; ++at) {
      int mloc = wm * 32 + at * 16 + ln;
      af[at].q[0] = *(const u32x4*)&As[buf][mloc][hi * 8];
      af[at].q[1] = *(const u32x4*)&As[buf][mloc][16 + hi * 8];
    }
#pragma unroll
    for (int bt = 0; bt < 4; ++bt) {
      Frag16 bfr;
      int nloc = wn * 64 + bt * 16 + ln;
      bfr.q[0] = *(const u32x4*)&Bs[buf][nloc][hi * 8];
      bfr.q[1] = *(const u32x4*)&Bs[buf][nloc][16 + hi * 8];
      WMMA_BF16(acc[0][bt].v, af[0].v, bfr.v);
      WMMA_BF16(acc[1][bt].v, af[1].v, bfr.v);
    }
  }
#undef STAGE_TILE

#pragma unroll
  for (int at = 0; at < 2; ++at)
#pragma unroll
    for (int bt = 0; bt < 4; ++bt) {
      int n = n0 + wn * 64 + bt * 16 + ln;
      float bv = bias ? bias[n] : 0.f;
#pragma unroll
      for (int j = 0; j < 8; ++j) {
        int m = m0 + wm * 32 + at * 16 + hi * 8 + j;  // C: vgpr j -> row j+8*hi
        float v = acc[at][bt].f[j] + bv;
        if (outF) outF[(size_t)m * N + n] = v;
        else      outB[(size_t)m * N + n] = f32_to_bf16(v);
      }
    }
}

// ---------------------------------------------------------------------------
// WMMA attention: per block = (b, h, 64 q rows). S=128 keys, d=96.
// S = QK^T*scale -> softmax (deferred norm) -> O = P V (normalized at store).
// ---------------------------------------------------------------------------
__global__ __launch_bounds__(256) void attention_kernel(
    const unsigned short* __restrict__ Qb,   // [B*P, E] bf16
    const unsigned short* __restrict__ Kb,   // [B*S, E] bf16
    const unsigned short* __restrict__ Vb,   // [B*S, E] bf16
    unsigned short* __restrict__ Ob) {       // [B*P, E] bf16
  constexpr int Ec = 768, Dh = 96, Sk = 128, Pq = 1024;
  __shared__ alignas(16) unsigned short Ks[128][96];   // [key][d]
  __shared__ alignas(16) unsigned short Ps[64][128];   // probs bf16
  __shared__ alignas(16) unsigned short Vt[96][64];    // [d][key] chunk
  __shared__ float smax[2][64];
  __shared__ float ssum[2][64];
  __shared__ float sinv[64];

  int bx = blockIdx.x;
  int qt = bx & 15, h = (bx >> 4) & 7, b = bx >> 7;
  int tid = threadIdx.x;
  int w = tid >> 5, lane = tid & 31, hi = lane >> 4, ln = lane & 15;
  int rt = w & 3, ch = w >> 2;   // 4 row-tiles, 2 column halves

  // stage K head-slice [128 x 96] (async b128 copies)
  for (int it = tid; it < (Sk * Dh) / 8; it += 256) {
    int r = it / 12, sg = it % 12;
    stage_b128(Kb + ((size_t)(b * Sk + r)) * Ec + h * Dh + sg * 8,
               &Ks[r][sg * 8]);
  }
  wait_async_all();
  __syncthreads();

  // ---- S = Q K^T ----
  Acc8 sacc[4];
#pragma unroll
  for (int t = 0; t < 4; ++t)
#pragma unroll
    for (int j = 0; j < 8; ++j) sacc[t].f[j] = 0.f;

  int qrow = b * Pq + qt * 64 + rt * 16 + ln;
  for (int ks = 0; ks < 3; ++ks) {       // d = 96 = 3 x 32
    Frag16 a;
    const unsigned short* qp = Qb + (size_t)qrow * Ec + h * Dh + ks * 32;
    a.q[0] = *(const u32x4*)(qp + hi * 8);
    a.q[1] = *(const u32x4*)(qp + 16 + hi * 8);
#pragma unroll
    for (int t = 0; t < 4; ++t) {
      Frag16 bfr;
      int key = ch * 64 + t * 16 + ln;
      bfr.q[0] = *(const u32x4*)&Ks[key][ks * 32 + hi * 8];
      bfr.q[1] = *(const u32x4*)&Ks[key][ks * 32 + 16 + hi * 8];
      WMMA_BF16(sacc[t].v, a.v, bfr.v);
    }
  }

  const float scale = 0.1020620726f;     // 1/sqrt(96)
  float pm[8];
#pragma unroll
  for (int j = 0; j < 8; ++j) {
    float m = -1e30f;
#pragma unroll
    for (int t = 0; t < 4; ++t) { sacc[t].f[j] *= scale; m = fmaxf(m, sacc[t].f[j]); }
    for (int off = 1; off < 16; off <<= 1) m = fmaxf(m, __shfl_xor(m, off, 16));
    pm[j] = m;
  }
  if (ln == 0)
#pragma unroll
    for (int j = 0; j < 8; ++j) smax[ch][rt * 16 + hi * 8 + j] = pm[j];
  __syncthreads();

  float psum[8];
#pragma unroll
  for (int j = 0; j < 8; ++j) {
    int r = rt * 16 + hi * 8 + j;
    float gm = fmaxf(smax[0][r], smax[1][r]);
    float s = 0.f;
#pragma unroll
    for (int t = 0; t < 4; ++t) {
      float p = __expf(sacc[t].f[j] - gm);
      sacc[t].f[j] = p;
      s += p;
    }
    for (int off = 1; off < 16; off <<= 1) s += __shfl_xor(s, off, 16);
    psum[j] = s;
  }
  if (ln == 0)
#pragma unroll
    for (int j = 0; j < 8; ++j) ssum[ch][rt * 16 + hi * 8 + j] = psum[j];
#pragma unroll
  for (int t = 0; t < 4; ++t)
#pragma unroll
    for (int j = 0; j < 8; ++j)
      Ps[rt * 16 + hi * 8 + j][ch * 64 + t * 16 + ln] = f32_to_bf16(sacc[t].f[j]);
  __syncthreads();
  if (tid < 64) sinv[tid] = 1.f / (ssum[0][tid] + ssum[1][tid] + 1e-20f);

  // ---- O = P V ----  (V transposed in LDS, two 64-key chunks)
  int dset = ch;
  Acc8 oacc[3];
#pragma unroll
  for (int t = 0; t < 3; ++t)
#pragma unroll
    for (int j = 0; j < 8; ++j) oacc[t].f[j] = 0.f;

  for (int kc = 0; kc < 2; ++kc) {
    __syncthreads();
    for (int it = tid; it < Dh * 64; it += 256) {
      int d = it >> 6, k = it & 63;
      Vt[d][k] = Vb[((size_t)(b * Sk + kc * 64 + k)) * Ec + h * Dh + d];
    }
    __syncthreads();
#pragma unroll
    for (int ks2 = 0; ks2 < 2; ++ks2) {
      Frag16 a;
      int m = rt * 16 + ln, kb = kc * 64 + ks2 * 32;
      a.q[0] = *(const u32x4*)&Ps[m][kb + hi * 8];
      a.q[1] = *(const u32x4*)&Ps[m][kb + 16 + hi * 8];
#pragma unroll
      for (int td = 0; td < 3; ++td) {
        Frag16 bfr;
        int d = (dset * 3 + td) * 16 + ln;
        bfr.q[0] = *(const u32x4*)&Vt[d][ks2 * 32 + hi * 8];
        bfr.q[1] = *(const u32x4*)&Vt[d][ks2 * 32 + 16 + hi * 8];
        WMMA_BF16(oacc[td].v, a.v, bfr.v);
      }
    }
  }
#pragma unroll
  for (int td = 0; td < 3; ++td) {
    int dcol = (dset * 3 + td) * 16 + ln;
#pragma unroll
    for (int j = 0; j < 8; ++j) {
      int r = rt * 16 + hi * 8 + j;
      float val = oacc[td].f[j] * sinv[r];
      Ob[((size_t)(b * Pq + qt * 64 + r)) * Ec + h * Dh + dcol] = f32_to_bf16(val);
    }
  }
}

// ---------------------------------------------------------------------------
// grounded = LN(visual + fused) -> bf16    (one block per row)
// ---------------------------------------------------------------------------
__global__ __launch_bounds__(256) void grounded_ln_kernel(
    const float* __restrict__ visual, const unsigned short* __restrict__ fused,
    const float* __restrict__ gamma, const float* __restrict__ beta,
    unsigned short* __restrict__ out) {
  constexpr int Ec = 768;
  __shared__ float xb[Ec];
  __shared__ float red[256];
  int row = blockIdx.x, tid = threadIdx.x;
  float s = 0.f, s2 = 0.f;
  for (int e = tid; e < Ec; e += 256) {
    float v = visual[(size_t)row * Ec + e] + bf16_to_f32(fused[(size_t)row * Ec + e]);
    xb[e] = v; s += v; s2 += v * v;
  }
  float tot = block_sum(s, red);
  float tot2 = block_sum(s2, red);
  float mean = tot / Ec;
  float var = tot2 / Ec - mean * mean;
  float rstd = rsqrtf(var + 1e-5f);
  for (int e = tid; e < Ec; e += 256)
    out[(size_t)row * Ec + e] =
        f32_to_bf16((xb[e] - mean) * rstd * gamma[e] + beta[e]);
}

// ---------------------------------------------------------------------------
// Graph node update (batch independent): one block per class
// ---------------------------------------------------------------------------
__global__ __launch_bounds__(256) void graph_update_kernel(
    const float* __restrict__ node_emb, const float* __restrict__ edge_emb,
    const float* __restrict__ W1, const float* __restrict__ b1,
    const float* __restrict__ lng, const float* __restrict__ lnb,
    const float* __restrict__ W2, const float* __restrict__ b2,
    float* __restrict__ updated) {
  constexpr int Ec = 768;
  __shared__ float comb[2 * Ec];
  __shared__ float hbuf[Ec];
  __shared__ float red[256];
  int i = blockIdx.x, tid = threadIdx.x;
  const int ncnt[5] = {2, 2, 3, 1, 0};
  const int nbr[5][3] = {{1, 2, 0}, {0, 2, 0}, {0, 1, 3}, {2, 0, 0}, {0, 0, 0}};
  for (int e = tid; e < Ec; e += 256) {
    float ne = node_emb[(size_t)i * Ec + e];
    float s0 = 0.f;
    for (int t = 0; t < ncnt[i]; ++t) s0 += node_emb[(size_t)nbr[i][t] * Ec + e];
    float ef = s0 * edge_emb[e];                                  // r=0 co-occ
    if (i == 3) ef += node_emb[2 * Ec + e] * edge_emb[Ec + e];    // r=1 hier
    ef += ne * edge_emb[2 * Ec + e];                              // r=2 self
    comb[e] = ne;
    comb[Ec + e] = ef * (1.f / 3.f);
  }
  __syncthreads();
  for (int f = tid; f < Ec; f += 256) {
    const float* wr = W1 + (size_t)f * 2 * Ec;
    float a = b1[f];
    for (int e = 0; e < 2 * Ec; ++e) a += wr[e] * comb[e];
    hbuf[f] = a;
  }
  __syncthreads();
  float s = 0.f, s2 = 0.f;
  for (int f = tid; f < Ec; f += 256) { float v = hbuf[f]; s += v; s2 += v * v; }
  float tot = block_sum(s, red), tot2 = block_sum(s2, red);
  float mean = tot / Ec, var = tot2 / Ec - mean * mean, rstd = rsqrtf(var + 1e-5f);
  for (int f = tid; f < Ec; f += 256)
    comb[f] = fmaxf((hbuf[f] - mean) * rstd * lng[f] + lnb[f], 0.f);
  __syncthreads();
  for (int f = tid; f < Ec; f += 256) {
    const float* wr = W2 + (size_t)f * Ec;
    float a = b2[f];
    for (int e = 0; e < Ec; ++e) a += wr[e] * comb[e];
    updated[(size_t)i * Ec + f] = a;
  }
}

// ---------------------------------------------------------------------------
// Graph attention/scoring + sent update: one block per batch
// ---------------------------------------------------------------------------
__global__ __launch_bounds__(256) void graph_attn_kernel(
    const float* __restrict__ Wqkv, const float* __restrict__ bqkv,
    const float* __restrict__ updated, const float* __restrict__ presence,
    float* __restrict__ sent, float* __restrict__ act,
    float* __restrict__ scores_out) {
  constexpr int Ec = 768, NCc = 5, Hh = 8, Dh = 96;
  __shared__ float sb[Ec];
  __shared__ float qh[Ec];
  __shared__ float kh[NCc * Ec];
  __shared__ float sc[Hh][NCc];
  __shared__ float fsc[NCc];
  int b = blockIdx.x, tid = threadIdx.x;
  for (int e = tid; e < Ec; e += 256) sb[e] = sent[(size_t)b * Ec + e];
  __syncthreads();
  for (int f = tid; f < Ec; f += 256) {           // q = Wq sent + bq
    const float* wr = Wqkv + (size_t)f * Ec;
    float a = bqkv[f];
    for (int e = 0; e < Ec; ++e) a += wr[e] * sb[e];
    qh[f] = a;
  }
  for (int idx = tid; idx < NCc * Ec; idx += 256) {  // k = Wk updated + bk
    int c = idx / Ec, f = idx % Ec;
    const float* wr = Wqkv + (size_t)(Ec + f) * Ec;
    float a = bqkv[Ec + f];
    for (int e = 0; e < Ec; ++e) a += wr[e] * updated[(size_t)c * Ec + e];
    kh[idx] = a;
  }
  __syncthreads();
  if (tid < Hh * NCc) {
    int h = tid / NCc, c = tid % NCc;
    float s = 0.f;
    for (int d = 0; d < Dh; ++d) s += qh[h * Dh + d] * kh[c * Ec + h * Dh + d];
    sc[h][c] = s * 0.1020620726f;
  }
  __syncthreads();
  if (tid < Hh) {
    int h = tid;
    float m = -1e30f;
    for (int c = 0; c < NCc; ++c) m = fmaxf(m, sc[h][c]);
    float s = 0.f;
    for (int c = 0; c < NCc; ++c) { float p = __expf(sc[h][c] - m); sc[h][c] = p; s += p; }
    for (int c = 0; c < NCc; ++c) sc[h][c] /= s;
  }
  __syncthreads();
  if (tid == 0) {
    float wv[NCc];
    float m = -1e30f;
    for (int c = 0; c < NCc; ++c) {
      float a = 0.f;
      for (int h = 0; h < Hh; ++h) a += sc[h][c];
      wv[c] = (a / Hh) * (presence[(size_t)b * NCc + c] + 0.1f);
      m = fmaxf(m, wv[c]);
    }
    float s = 0.f;
    for (int c = 0; c < NCc; ++c) { float p = __expf(wv[c] - m); fsc[c] = p; s += p; }
    for (int c = 0; c < NCc; ++c) {
      fsc[c] /= s;
      if (scores_out) scores_out[(size_t)b * NCc + c] = fsc[c];
    }
  }
  __syncthreads();
  for (int f = tid; f < Ec; f += 256) {
    float accm = 0.f;
    for (int c = 0; c < NCc; ++c) {
      float a = updated[(size_t)c * Ec + f] * fsc[c];
      act[((size_t)b * NCc + c) * Ec + f] = a;
      accm += a;
    }
    sent[(size_t)b * Ec + f] = sb[f] + accm * (1.f / NCc);
  }
}

// ---------------------------------------------------------------------------
// Per-class MLPs + global fusion: one block per (b, c)
// ---------------------------------------------------------------------------
__global__ __launch_bounds__(256) void class_guidance_kernel(
    const float* __restrict__ act, const float* __restrict__ sent,
    const float* __restrict__ cW1, const float* __restrict__ cb1,
    const float* __restrict__ clng, const float* __restrict__ clnb,
    const float* __restrict__ cW2, const float* __restrict__ cb2,
    const float* __restrict__ gfW1, const float* __restrict__ gfb1,
    const float* __restrict__ gflng, const float* __restrict__ gflnb,
    const float* __restrict__ gfW2, const float* __restrict__ gfb2,
    float* __restrict__ outCG) {
  constexpr int Ec = 768, NCc = 5;
  __shared__ float x[Ec];
  __shared__ float h1[Ec];
  __shared__ float gin[2 * Ec];
  __shared__ float red[256];
  int bc = blockIdx.x, tid = threadIdx.x;
  int b = bc / NCc, c = bc % NCc;
  for (int e = tid; e < Ec; e += 256) x[e] = act[(size_t)bc * Ec + e];
  __syncthreads();
  for (int f = tid; f < Ec; f += 256) {
    const float* wr = cW1 + ((size_t)c * Ec + f) * Ec;
    float a = cb1[c * Ec + f];
    for (int e = 0; e < Ec; ++e) a += wr[e] * x[e];
    h1[f] = a;
  }
  __syncthreads();
  float s = 0.f, s2 = 0.f;
  for (int f = tid; f < Ec; f += 256) { float v = h1[f]; s += v; s2 += v * v; }
  float tot = block_sum(s, red), tot2 = block_sum(s2, red);
  float mean = tot / Ec, var = tot2 / Ec - mean * mean, rstd = rsqrtf(var + 1e-5f);
  for (int f = tid; f < Ec; f += 256)
    x[f] = fmaxf((h1[f] - mean) * rstd * clng[c * Ec + f] + clnb[c * Ec + f], 0.f);
  __syncthreads();
  for (int f = tid; f < Ec; f += 256) {
    const float* wr = cW2 + ((size_t)c * Ec + f) * Ec;
    float a = cb2[c * Ec + f];
    for (int e = 0; e < Ec; ++e) a += wr[e] * x[e];
    gin[f] = a;
  }
  for (int f = tid; f < Ec; f += 256) gin[Ec + f] = sent[(size_t)b * Ec + f];
  __syncthreads();
  for (int f = tid; f < Ec; f += 256) {
    const float* wr = gfW1 + (size_t)f * 2 * Ec;
    float a = gfb1[f];
    for (int e = 0; e < 2 * Ec; ++e) a += wr[e] * gin[e];
    h1[f] = a;
  }
  __syncthreads();
  s = 0.f; s2 = 0.f;
  for (int f = tid; f < Ec; f += 256) { float v = h1[f]; s += v; s2 += v * v; }
  tot = block_sum(s, red); tot2 = block_sum(s2, red);
  mean = tot / Ec; var = tot2 / Ec - mean * mean; rstd = rsqrtf(var + 1e-5f);
  for (int f = tid; f < Ec; f += 256)
    x[f] = fmaxf((h1[f] - mean) * rstd * gflng[f] + gflnb[f], 0.f);
  __syncthreads();
  for (int f = tid; f < Ec; f += 256) {
    const float* wr = gfW2 + (size_t)f * Ec;
    float a = gfb2[f];
    for (int e = 0; e < Ec; ++e) a += wr[e] * x[e];
    outCG[(size_t)bc * Ec + f] = a;
  }
}

// ---------------------------------------------------------------------------
extern "C" void kernel_launch(void* const* d_in, const int* in_sizes, int n_in,
                              void* d_out, int out_size, void* d_ws, size_t ws_size,
                              hipStream_t stream) {
  constexpr int Bc = 32, Pp = 1024, Ss = 128, Ec = 768, NCc = 5, NLc = 3;

  const float* visual   = (const float*)d_in[0];
  const float* text     = (const float*)d_in[1];
  const float* sent_in  = (const float*)d_in[2];
  const float* presence = (const float*)d_in[3];
  const float* g_node   = (const float*)d_in[4];
  const float* g_edge   = (const float*)d_in[5];
  const float* g_Wqkv   = (const float*)d_in[6];
  const float* g_bqkv   = (const float*)d_in[7];
  const float* g_W1     = (const float*)d_in[10];
  const float* g_b1     = (const float*)d_in[11];
  const float* g_lng    = (const float*)d_in[12];
  const float* g_lnb    = (const float*)d_in[13];
  const float* g_W2     = (const float*)d_in[14];
  const float* g_b2     = (const float*)d_in[15];
  const float* v_Wqkv   = (const float*)d_in[16];
  const float* v_bqkv   = (const float*)d_in[17];
  const float* v_Wo     = (const float*)d_in[18];
  const float* v_bo     = (const float*)d_in[19];
  const float* c_W1     = (const float*)d_in[20];
  const float* c_b1     = (const float*)d_in[21];
  const float* c_lng    = (const float*)d_in[22];
  const float* c_lnb    = (const float*)d_in[23];
  const float* c_W2     = (const float*)d_in[24];
  const float* c_b2     = (const float*)d_in[25];
  const float* gf_W1    = (const float*)d_in[26];
  const float* gf_b1    = (const float*)d_in[27];
  const float* gf_lng   = (const float*)d_in[28];
  const float* gf_lnb   = (const float*)d_in[29];
  const float* gf_W2    = (const float*)d_in[30];
  const float* gf_b2    = (const float*)d_in[31];
  const float* out_W    = (const float*)d_in[32];
  const float* out_b    = (const float*)d_in[33];
  const float* norm_g   = (const float*)d_in[34];
  const float* norm_b   = (const float*)d_in[35];

  float* outGO = (float*)d_out;                          // [B,P,E]
  float* outCG = outGO + (size_t)Bc * Pp * Ec;           // [B,NC,E]
  float* outSC = outCG + (size_t)Bc * NCc * Ec;          // [B,NC]

  unsigned char* base = (unsigned char*)d_ws;
  size_t off = 0;
  auto alloc = [&](size_t bytes) -> void* {
    void* p = base + off;
    off = (off + bytes + 255) & ~(size_t)255;
    return p;
  };
  const size_t MV = (size_t)Bc * Pp;   // 32768 visual rows
  const size_t MT = (size_t)Bc * Ss;   // 4096 text rows

  unsigned short* V16    = (unsigned short*)alloc(MV * Ec * 2);
  unsigned short* T16    = (unsigned short*)alloc(MT * Ec * 2);
  unsigned short* Wqkv16 = (unsigned short*)alloc((size_t)3 * Ec * Ec * 2);
  unsigned short* Wo16   = (unsigned short*)alloc((size_t)Ec * Ec * 2);
  unsigned short* Wout16 = (unsigned short*)alloc((size_t)Ec * Ec * 2);
  unsigned short* Qb     = (unsigned short*)alloc(MV * Ec * 2);
  unsigned short* Kb     = (unsigned short*)alloc(MT * Ec * 2);
  unsigned short* Vb     = (unsigned short*)alloc(MT * Ec * 2);
  unsigned short* Ob     = (unsigned short*)alloc(MV * Ec * 2);
  float* sent    = (float*)alloc((size_t)Bc * Ec * 4);
  float* updated = (float*)alloc((size_t)NCc * Ec * 4);
  float* act     = (float*)alloc((size_t)Bc * NCc * Ec * 4);
  unsigned short* Fusedb = Qb;   // Q dead after attention
  unsigned short* G16    = V16;  // visual-bf16 dead after Q GEMM

  dim3 blk(256);

  // fp32 -> bf16 staging
  cvt_f32_bf16_kernel<<<2048, blk, 0, stream>>>(visual, V16, (int)(MV * Ec));
  cvt_f32_bf16_kernel<<<512, blk, 0, stream>>>(text, T16, (int)(MT * Ec));
  cvt_f32_bf16_kernel<<<256, blk, 0, stream>>>(v_Wqkv, Wqkv16, 3 * Ec * Ec);
  cvt_f32_bf16_kernel<<<128, blk, 0, stream>>>(v_Wo, Wo16, Ec * Ec);
  cvt_f32_bf16_kernel<<<128, blk, 0, stream>>>(out_W, Wout16, Ec * Ec);

  // graph reasoning (node path is batch-independent -> computed once/layer)
  (void)hipMemcpyAsync(sent, sent_in, (size_t)Bc * Ec * 4,
                       hipMemcpyDeviceToDevice, stream);
  for (int l = 0; l < NLc; ++l) {
    graph_update_kernel<<<NCc, blk, 0, stream>>>(
        g_node + (size_t)l * NCc * Ec, g_edge + (size_t)l * 3 * Ec,
        g_W1 + (size_t)l * Ec * 2 * Ec, g_b1 + (size_t)l * Ec,
        g_lng + (size_t)l * Ec, g_lnb + (size_t)l * Ec,
        g_W2 + (size_t)l * Ec * Ec, g_b2 + (size_t)l * Ec, updated);
    graph_attn_kernel<<<Bc, blk, 0, stream>>>(
        g_Wqkv + (size_t)l * 3 * Ec * Ec, g_bqkv + (size_t)l * 3 * Ec,
        updated, presence, sent, act,
        (l == NLc - 1) ? outSC : (float*)nullptr);
  }

  // visual-text fusion MHA (WMMA path), 128x128 block tiles
  gemm_bf16_kernel<<<dim3(MV / 128, Ec / 128), blk, 0, stream>>>(
      V16, Wqkv16, v_bqkv, nullptr, Qb, (int)MV, Ec, Ec);
  gemm_bf16_kernel<<<dim3(MT / 128, Ec / 128), blk, 0, stream>>>(
      T16, Wqkv16 + (size_t)Ec * Ec, v_bqkv + Ec, nullptr, Kb, (int)MT, Ec, Ec);
  gemm_bf16_kernel<<<dim3(MT / 128, Ec / 128), blk, 0, stream>>>(
      T16, Wqkv16 + (size_t)2 * Ec * Ec, v_bqkv + 2 * Ec, nullptr, Vb, (int)MT, Ec, Ec);
  attention_kernel<<<Bc * 8 * 16, blk, 0, stream>>>(Qb, Kb, Vb, Ob);
  gemm_bf16_kernel<<<dim3(MV / 128, Ec / 128), blk, 0, stream>>>(
      Ob, Wo16, v_bo, nullptr, Fusedb, (int)MV, Ec, Ec);
  grounded_ln_kernel<<<(int)MV, blk, 0, stream>>>(visual, Fusedb, norm_g, norm_b, G16);
  gemm_bf16_kernel<<<dim3(MV / 128, Ec / 128), blk, 0, stream>>>(
      G16, Wout16, out_b, outGO, nullptr, (int)MV, Ec, Ec);

  // class guidance head
  class_guidance_kernel<<<Bc * NCc, blk, 0, stream>>>(
      act, sent, c_W1, c_b1, c_lng, c_lnb, c_W2, c_b2,
      gf_W1, gf_b1, gf_lng, gf_lnb, gf_W2, gf_b2, outCG);

  (void)in_sizes; (void)n_in; (void)out_size; (void)ws_size;
}
